// LongRangeTransitionLayer_78941498900524
// MI455X (gfx1250) — matrix operand order repaired and checked by reading, not compiled, not run on hardware
//
#include <hip/hip_runtime.h>
#include <math.h>

// LongRangeTransitionLayer fused kernel for gfx1250 (MI455X)
// GEMM [16384 x 512] x [512 x 288] via v_wmma_f32_16x16x32_f16.
// Weights staged in LDS in *fragment-major* f16 layout: one 1KB block per
// (k-step, n-tile), [half][lane][8 halves], so the inner loop reads B frags
// with immediate DS offsets and conflict-free bank access.
// Fused 4x4 ortho-exp epilogue reads the projection tile back from LDS.

#define TD         4
#define NH         8
#define IDIM       512
#define NTOT       288      // 128 in_pre + 128 out_pre + 32 eig
#define PSTRIDE    296      // padded floats per token row of P
#define TOK_PER_WG 128
#define KSTEPS     16       // 512 / 32
#define NTILES     18       // 288 / 16
#define LDS_BYTES  (KSTEPS * NTILES * 1024)   // 294,912 B <= 320 KB

typedef __attribute__((ext_vector_type(16))) _Float16 v16h;
typedef __attribute__((ext_vector_type(8)))  _Float16 v8h;
typedef __attribute__((ext_vector_type(8)))  float    v8f;

__device__ __forceinline__ void mm4(const float* A, const float* B, float* C) {
#pragma unroll
    for (int a = 0; a < 4; ++a)
#pragma unroll
        for (int c = 0; c < 4; ++c) {
            float s = 0.f;
#pragma unroll
            for (int b = 0; b < 4; ++b) s = fmaf(A[a * 4 + b], B[b * 4 + c], s);
            C[a * 4 + c] = s;
        }
}

// exp(0.1*(M - M^T)) via normalizer-scaled order-16 Taylor series (matches ref)
__device__ __forceinline__ void ortho_exp4(const float* pre, float* outm) {
    float sk[16];
    float ss = 0.f;
#pragma unroll
    for (int a = 0; a < 4; ++a)
#pragma unroll
        for (int c = 0; c < 4; ++c) {
            float s = 0.1f * (pre[a * 4 + c] - pre[c * 4 + a]);
            sk[a * 4 + c] = s;
            ss += s * s + 1e-3f;
        }
    float norm = fmaxf(sqrtf(ss), 1.0f);
    float inv = 1.0f / norm;
#pragma unroll
    for (int i = 0; i < 16; ++i) sk[i] *= inv;

    float term[16], acc[16];
#pragma unroll
    for (int i = 0; i < 16; ++i) {
        float e = ((i & 3) == (i >> 2)) ? 1.f : 0.f;
        term[i] = e;
        acc[i] = e;
    }
#pragma unroll 1
    for (int o = 1; o < 16; ++o) {
        float nt[16];
        mm4(term, sk, nt);
        float c = norm / (float)o;
#pragma unroll
        for (int i = 0; i < 16; ++i) {
            term[i] = nt[i] * c;
            acc[i] += term[i];
        }
    }
#pragma unroll
    for (int i = 0; i < 16; ++i) outm[i] = acc[i];
}

__global__ __launch_bounds__(256) void lrt_fused_kernel(
    const float* __restrict__ x,
    const float* __restrict__ inW, const float* __restrict__ outW,
    const float* __restrict__ eigW,
    const float* __restrict__ inB, const float* __restrict__ outB,
    const float* __restrict__ eigB,
    float* __restrict__ out)
{
    extern __shared__ char smem[];
    _Float16* ldsW = (_Float16*)smem;   // phase 0/1: f16 weights, frag-major
    float*    ldsP = (float*)smem;      // phase 2: projections [128][296]

    const int tid    = threadIdx.x;
    const int lane   = tid & 31;
    const int wave   = tid >> 5;
    const int row    = lane & 15;       // M row (A) / N col (B) within tile
    const int laneHi = lane >> 4;       // 0 or 1
    const int wg     = blockIdx.x;

    // ---- Phase 0: stage combined weight matrix [288][512] f32 -> f16 in LDS,
    // rearranged into B-fragment order.
    // Fragment block (kt, nt) = 1024 B at ((kt*18+nt)<<10).
    // Within block: half h (K 0-7 / 8-15 within lane), lane L, elem e:
    //   B elem (L, e) corresponds to K = e + 16*(L>=16), N = L%16
    //   byte = block*1024 + h*512 + L*16 + (e&7)*2
#pragma unroll 1
    for (int i = 0; i < 144; ++i) {
        int g = tid + i * 256;          // float4 chunk index (36864 total)
        int n = g >> 7;                 // weight row 0..287
        int k = (g & 127) << 2;         // 0..508 (chunk of 4 k's)
        const float* src;
        if (n < 128)      src = inW  + ((size_t)n << 9);
        else if (n < 256) src = outW + ((size_t)(n - 128) << 9);
        else              src = eigW + ((size_t)(n - 256) << 9);
        float4 v = *(const float4*)(src + k);

        int nt = n >> 4;
        int kt = k >> 5;
        int L  = (n & 15) + (((k >> 4) & 1) << 4);
        int e  = k & 15;                // e..e+3, stays within one 8-elem half
        int h  = e >> 3;
        // offset in halves: block*512 + h*256 + L*8 + (e&7)
        _Float16* dst = ldsW + ((kt * NTILES + nt) << 9) + (h << 8) + (L << 3) + (e & 7);
        dst[0] = (_Float16)v.x;
        dst[1] = (_Float16)v.y;
        dst[2] = (_Float16)v.z;
        dst[3] = (_Float16)v.w;
    }
    __syncthreads();

    // ---- Phase 1: WMMA GEMM. Each wave: 16 tokens x 288 outputs.
    v8f acc[NTILES];
#pragma unroll
    for (int t = 0; t < NTILES; ++t)
#pragma unroll
        for (int j = 0; j < 8; ++j) acc[t][j] = 0.f;

    const int tbase = wg * TOK_PER_WG + wave * 16;
    const float* xrow = x + ((size_t)(tbase + row) << 9);

#pragma unroll 1
    for (int kt = 0; kt < KSTEPS; ++kt) {
        // A fragment 16x32 f16: lanes 0-15 -> K 0-15, lanes 16-31 -> K 16-31
        int k0 = kt * 32 + laneHi * 16;
        float4 a0 = *(const float4*)(xrow + k0);
        float4 a1 = *(const float4*)(xrow + k0 + 4);
        float4 a2 = *(const float4*)(xrow + k0 + 8);
        float4 a3 = *(const float4*)(xrow + k0 + 12);
        v16h A;
        A[0]  = (_Float16)a0.x; A[1]  = (_Float16)a0.y;
        A[2]  = (_Float16)a0.z; A[3]  = (_Float16)a0.w;
        A[4]  = (_Float16)a1.x; A[5]  = (_Float16)a1.y;
        A[6]  = (_Float16)a1.z; A[7]  = (_Float16)a1.w;
        A[8]  = (_Float16)a2.x; A[9]  = (_Float16)a2.y;
        A[10] = (_Float16)a2.z; A[11] = (_Float16)a2.w;
        A[12] = (_Float16)a3.x; A[13] = (_Float16)a3.y;
        A[14] = (_Float16)a3.z; A[15] = (_Float16)a3.w;

        // B fragments: single per-kt base, all 18 tiles at immediate offsets
        const _Float16* bbase = ldsW + ((kt * NTILES) << 9) + (lane << 3);
#pragma unroll
        for (int t = 0; t < NTILES; ++t) {
            v8h blo = *(const v8h*)(bbase + (t << 9));         // +t*1024 B
            v8h bhi = *(const v8h*)(bbase + (t << 9) + 256);   // +512 B
            v16h Bm;
#pragma unroll
            for (int e = 0; e < 8; ++e) { Bm[e] = blo[e]; Bm[8 + e] = bhi[e]; }
            acc[t] = __builtin_amdgcn_wmma_f32_16x16x32_f16(
                false, A, false, Bm, (short)0, acc[t], false, false);
        }
    }
    __syncthreads();   // done reading ldsW; safe to overwrite with P

    // ---- Write P tile to LDS: token-local [128][296]
    // C/D layout: vgpr r, lane L -> M = r + 8*laneHi, N = L%16
#pragma unroll
    for (int t = 0; t < NTILES; ++t) {
        int n = t * 16 + row;
#pragma unroll
        for (int r = 0; r < 8; ++r) {
            int tl = wave * 16 + laneHi * 8 + r;
            ldsP[tl * PSTRIDE + n] = acc[t][r];
        }
    }
    __syncthreads();

    // ---- Phase 2: per (token, head) ortho-exp epilogue. 1024 pairs / 256 thr.
#pragma unroll 1
    for (int i = 0; i < 4; ++i) {
        int pair = tid + i * 256;
        int tl = pair >> 3;
        int h  = pair & 7;
        const float* P = ldsP + tl * PSTRIDE;

        float inp[16], outp[16];
#pragma unroll
        for (int e = 0; e < 16; ++e) {
            inp[e]  = P[h * 16 + e]       + inB[h * 16 + e];
            outp[e] = P[128 + h * 16 + e] + outB[h * 16 + e];
        }
        float eig[4];
#pragma unroll
        for (int b = 0; b < 4; ++b)
            eig[b] = tanhf(P[256 + h * 4 + b] + eigB[h * 4 + b]);

        float inM[16], outM[16];
        ortho_exp4(inp, inM);
        ortho_exp4(outp, outM);

        // T = outM @ diag(eig) @ inM
        float T[16];
#pragma unroll
        for (int a = 0; a < 4; ++a)
#pragma unroll
            for (int c = 0; c < 4; ++c) {
                float s = 0.f;
#pragma unroll
                for (int b = 0; b < 4; ++b)
                    s = fmaf(outM[a * 4 + b] * eig[b], inM[b * 4 + c], s);
                T[a * 4 + c] = s;
            }

        size_t token = (size_t)wg * TOK_PER_WG + tl;
        float* dst = out + (token * NH + h) * 16;
#pragma unroll
        for (int q = 0; q < 4; ++q) {
            float4 v = make_float4(T[q * 4 + 0], T[q * 4 + 1], T[q * 4 + 2], T[q * 4 + 3]);
            *(float4*)(dst + q * 4) = v;
        }
    }
}

extern "C" void kernel_launch(void* const* d_in, const int* in_sizes, int n_in,
                              void* d_out, int out_size, void* d_ws, size_t ws_size,
                              hipStream_t stream) {
    const float* x    = (const float*)d_in[0];
    const float* inB  = (const float*)d_in[1];
    const float* inW  = (const float*)d_in[2];
    const float* outB = (const float*)d_in[3];
    const float* outW = (const float*)d_in[4];
    const float* eigB = (const float*)d_in[5];
    const float* eigW = (const float*)d_in[6];
    float* out = (float*)d_out;

    int tokens = in_sizes[0] / IDIM;          // 16384
    int blocks = tokens / TOK_PER_WG;         // 128

    lrt_fused_kernel<<<dim3(blocks), dim3(256), LDS_BYTES, stream>>>(
        x, inW, outW, eigW, inB, outB, eigB, out);
}
